// TransformerDecoder_72679436583417
// MI455X (gfx1250) — matrix operand order repaired and checked by reading, compiled, not run on hardware
//
#include <hip/hip_runtime.h>
#include <hip/hip_bf16.h>

// ---------------- problem constants ----------------
#define B_    2
#define ST_   1024
#define SS_   1024
#define D_    512
#define H_    8
#define L_    4
#define DFF_  2048
#define V_    32000
#define DK_   64
#define NT_   (B_*ST_)                 // 2048 rows of activations
#define BHSS_ ((size_t)B_*H_*ST_*ST_)  // one attention tensor per layer

typedef __attribute__((ext_vector_type(16))) __bf16 v16bf_t;
typedef __attribute__((ext_vector_type(8)))  __bf16 v8bf_t;
typedef __attribute__((ext_vector_type(8)))  float  v8f_t;

// ---------------- elementwise kernels ----------------

// x = emb[tok]*sqrt(D) + posenc ; writes f32 + bf16
__global__ void embed_pe_kernel(const int* __restrict__ tokens,
                                const float* __restrict__ emb,
                                float* __restrict__ x, __bf16* __restrict__ xb) {
  int idx = blockIdx.x * 256 + threadIdx.x;          // over NT_*D_
  int d   = idx & (D_ - 1);
  int bt  = idx / D_;
  int t   = bt & (ST_ - 1);
  int tok = tokens[bt];
  float div = __expf(-(float)((d >> 1) << 1) * (9.210340371976184f / (float)D_));
  float ang = (float)t * div;
  float pe  = (d & 1) ? __cosf(ang) : __sinf(ang);
  float v   = emb[(size_t)tok * D_ + d] * 22.62741699796952f + pe; // sqrt(512)
  x[idx]  = v;
  xb[idx] = (__bf16)v;
}

__global__ void to_bf16_kernel(const float* __restrict__ in, __bf16* __restrict__ out, int n) {
  int i = blockIdx.x * 256 + threadIdx.x;
  if (i < n) out[i] = (__bf16)in[i];
}

// in: [K,N] f32 ; out: [N,K] bf16 ; batch of matrices via blockIdx.y
__global__ void transpose_bf16_kernel(const float* __restrict__ in, __bf16* __restrict__ out,
                                      int K, int N) {
  size_t mat = (size_t)blockIdx.y * (size_t)K * (size_t)N;
  int idx = blockIdx.x * 256 + threadIdx.x;
  if (idx >= K * N) return;
  int n = idx / K, k = idx - n * K;                  // coalesced writes
  out[mat + idx] = (__bf16)in[mat + (size_t)k * N + n];
}

// Vb: [B,S,D] bf16 -> VT: [B,H,DK,S] bf16
__global__ void transpose_v_kernel(const __bf16* __restrict__ Vb, __bf16* __restrict__ VT, int S) {
  int idx = blockIdx.x * 256 + threadIdx.x;          // over B*H*DK*S
  int s = idx % S; int r = idx / S;
  int d = r % DK_; r /= DK_;
  int h = r % H_;  int b = r / H_;
  VT[idx] = Vb[((size_t)b * S + s) * D_ + h * DK_ + d];
}

// row softmax in place on f32 scores + bf16 copy; causal masks keys > q
__global__ void softmax_kernel(float* __restrict__ S, __bf16* __restrict__ Pbf,
                               int Ncols, int causal) {
  int row = blockIdx.x;                               // B*H*ST rows
  int q   = row & (ST_ - 1);
  int valid = causal ? (q + 1) : Ncols;
  float* p = S + (size_t)row * Ncols;
  __bf16* pb = Pbf + (size_t)row * Ncols;
  __shared__ float red[256];
  int tid = threadIdx.x;
  float mx = -1e30f;
  for (int i = tid; i < valid; i += 256) mx = fmaxf(mx, p[i]);
  red[tid] = mx; __syncthreads();
  for (int s = 128; s > 0; s >>= 1) { if (tid < s) red[tid] = fmaxf(red[tid], red[tid + s]); __syncthreads(); }
  mx = red[0]; __syncthreads();
  float sum = 0.f;
  for (int i = tid; i < valid; i += 256) sum += __expf(p[i] - mx);
  red[tid] = sum; __syncthreads();
  for (int s = 128; s > 0; s >>= 1) { if (tid < s) red[tid] += red[tid + s]; __syncthreads(); }
  float inv = 1.0f / red[0];
  for (int i = tid; i < Ncols; i += 256) {
    float v = (i < valid) ? __expf(p[i] - mx) * inv : 0.0f;
    p[i] = v; pb[i] = (__bf16)v;
  }
}

// x = LayerNorm(x + res)*g + b ; biased variance ; writes f32 (in place) + bf16
__global__ void add_ln_kernel(float* __restrict__ x, __bf16* __restrict__ xb,
                              const float* __restrict__ res,
                              const float* __restrict__ g, const float* __restrict__ b) {
  int row = blockIdx.x; int tid = threadIdx.x;       // 256 threads, D_=512
  float* xr = x + (size_t)row * D_;
  const float* rr = res + (size_t)row * D_;
  float v0 = xr[tid]       + rr[tid];
  float v1 = xr[tid + 256] + rr[tid + 256];
  __shared__ float s1[256], s2[256];
  s1[tid] = v0 + v1; s2[tid] = v0 * v0 + v1 * v1;
  __syncthreads();
  for (int s = 128; s > 0; s >>= 1) { if (tid < s) { s1[tid] += s1[tid + s]; s2[tid] += s2[tid + s]; } __syncthreads(); }
  float mean = s1[0] * (1.0f / D_);
  float var  = s2[0] * (1.0f / D_) - mean * mean;
  float rstd = rsqrtf(var + 1e-5f);
  float y0 = (v0 - mean) * rstd * g[tid]       + b[tid];
  float y1 = (v1 - mean) * rstd * g[tid + 256] + b[tid + 256];
  xr[tid] = y0; xr[tid + 256] = y1;
  xb[(size_t)row * D_ + tid]       = (__bf16)y0;
  xb[(size_t)row * D_ + tid + 256] = (__bf16)y1;
}

// ---------------- WMMA bf16 GEMM ----------------
// C[M,N] = scale * (A[M,K] x B[K,N]) + bias ; B supplied transposed as BT[N,K].
// One wave computes a 32x64 block: 2 A fragments x 4 B fragments -> 8 WMMA
// tiles per K-step (12 b128 loads / 8 wmma). Explicit double-buffered
// software pipeline: fragment set f1 for step kk+32 is loaded before the
// WMMAs consume set f0 for step kk, so loads never alias WMMA sources
// (no hazard NOPs) and waits stay partial. All K here are multiples of 64.
// Batched over blockIdx.y: z -> (bb=z/Hdim, hh=z%Hdim) with independent strides.
__global__ __launch_bounds__(256, 1)
void gemm_wmma_bf16(const __bf16* __restrict__ A, int lda, long long sAb, long long sAh,
                    const __bf16* __restrict__ BT, int ldb, long long sBb, long long sBh,
                    float* __restrict__ Cf, __bf16* __restrict__ Cbf,
                    int ldc, long long sCb, long long sCh,
                    const float* __restrict__ bias,
                    int M, int N, int K, int Hdim, float scale, int relu) {
  int z  = blockIdx.y;
  int bb = z / Hdim, hh = z - bb * Hdim;
  const __bf16* Ab = A  + bb * sAb + hh * sAh;
  const __bf16* Bb = BT + bb * sBb + hh * sBh;
  size_t coff = (size_t)(bb * sCb + hh * sCh);

  int wid   = blockIdx.x * 8 + (threadIdx.x >> 5);
  int tilesN = N >> 6;
  if (wid >= (M >> 5) * tilesN) return;
  int tm = (wid / tilesN) << 5;
  int tn = (wid % tilesN) << 6;

  int lane = threadIdx.x & 31;
  int col  = lane & 15;
  int half = lane >> 4;

  // A fragment rows: lane&15 is M; lane-half selects K {0..7,16..23} vs {8..15,24..31}
  const __bf16* arow0 = Ab + (size_t)(tm + col) * lda + half * 8;
  const __bf16* arow1 = arow0 + (size_t)16 * lda;
  // B fragment: lane&15 is N; lane-half selects K {0..7,16..23} vs {8..15,24..31}
  const __bf16* br0 = Bb + (size_t)(tn + col) * ldb + half * 8;
  const __bf16* br1 = br0 + (size_t)16 * ldb;
  const __bf16* br2 = br0 + (size_t)32 * ldb;
  const __bf16* br3 = br0 + (size_t)48 * ldb;

  v8f_t acc00 = {}, acc01 = {}, acc02 = {}, acc03 = {};
  v8f_t acc10 = {}, acc11 = {}, acc12 = {}, acc13 = {};

  v16bf_t f0a0, f0a1, f0b0, f0b1, f0b2, f0b3;
  v16bf_t f1a0, f1a1, f1b0, f1b1, f1b2, f1b3;

#define LOADF(P, KO)                                                           \
  { ((v8bf_t*)&P##a0)[0] = *(const v8bf_t*)(arow0 + (KO));                     \
    ((v8bf_t*)&P##a0)[1] = *(const v8bf_t*)(arow0 + (KO) + 16);                \
    ((v8bf_t*)&P##a1)[0] = *(const v8bf_t*)(arow1 + (KO));                     \
    ((v8bf_t*)&P##a1)[1] = *(const v8bf_t*)(arow1 + (KO) + 16);                \
    ((v8bf_t*)&P##b0)[0] = *(const v8bf_t*)(br0 + (KO));                       \
    ((v8bf_t*)&P##b0)[1] = *(const v8bf_t*)(br0 + (KO) + 16);                  \
    ((v8bf_t*)&P##b1)[0] = *(const v8bf_t*)(br1 + (KO));                       \
    ((v8bf_t*)&P##b1)[1] = *(const v8bf_t*)(br1 + (KO) + 16);                  \
    ((v8bf_t*)&P##b2)[0] = *(const v8bf_t*)(br2 + (KO));                       \
    ((v8bf_t*)&P##b2)[1] = *(const v8bf_t*)(br2 + (KO) + 16);                  \
    ((v8bf_t*)&P##b3)[0] = *(const v8bf_t*)(br3 + (KO));                       \
    ((v8bf_t*)&P##b3)[1] = *(const v8bf_t*)(br3 + (KO) + 16); }

#define WMMA_(AF, BF, ACC)                                                     \
  ACC = __builtin_amdgcn_wmma_f32_16x16x32_bf16(false, AF, false, BF,          \
                                                (short)0, ACC, false, false);
#define COMPF(P)                                                               \
  { WMMA_(P##a0, P##b0, acc00) WMMA_(P##a1, P##b0, acc10)                      \
    WMMA_(P##a0, P##b1, acc01) WMMA_(P##a1, P##b1, acc11)                      \
    WMMA_(P##a0, P##b2, acc02) WMMA_(P##a1, P##b2, acc12)                      \
    WMMA_(P##a0, P##b3, acc03) WMMA_(P##a1, P##b3, acc13) }

  LOADF(f0, 0);
  int kk = 0;
  for (; kk + 64 < K; kk += 64) {
    LOADF(f1, kk + 32); COMPF(f0);
    LOADF(f0, kk + 64); COMPF(f1);
  }
  LOADF(f1, kk + 32); COMPF(f0); COMPF(f1);
#undef LOADF
#undef WMMA_
#undef COMPF

#define EPILOG(ACC, MOFF, JJ)                                                  \
  { int n = tn + (JJ) * 16 + col;                                              \
    float bv = bias ? bias[n] : 0.0f;                                          \
    _Pragma("unroll") for (int r = 0; r < 8; ++r) {                            \
      int m = tm + (MOFF) + half * 8 + r;                                      \
      float val = ACC[r] * scale + bv;                                         \
      if (relu) val = fmaxf(val, 0.0f);                                        \
      size_t o = coff + (size_t)m * ldc + n;                                   \
      if (Cf)  Cf[o]  = val;                                                   \
      if (Cbf) Cbf[o] = (__bf16)val; } }
  EPILOG(acc00, 0, 0) EPILOG(acc01, 0, 1) EPILOG(acc02, 0, 2) EPILOG(acc03, 0, 3)
  EPILOG(acc10, 16, 0) EPILOG(acc11, 16, 1) EPILOG(acc12, 16, 2) EPILOG(acc13, 16, 3)
#undef EPILOG
}

// ---------------- host side ----------------
static void launch_gemm(hipStream_t st,
                        const __bf16* A, int lda, long long sAb, long long sAh,
                        const __bf16* BT, int ldb, long long sBb, long long sBh,
                        float* Cf, __bf16* Cbf, int ldc, long long sCb, long long sCh,
                        const float* bias, int M, int N, int K, int batch, int Hdim,
                        float scale, int relu) {
  int tiles = (M / 32) * (N / 64);
  dim3 grid((tiles + 7) / 8, batch);
  gemm_wmma_bf16<<<grid, 256, 0, st>>>(A, lda, sAb, sAh, BT, ldb, sBb, sBh,
                                       Cf, Cbf, ldc, sCb, sCh, bias,
                                       M, N, K, Hdim, scale, relu);
}

extern "C" void kernel_launch(void* const* d_in, const int* in_sizes, int n_in,
                              void* d_out, int out_size, void* d_ws, size_t ws_size,
                              hipStream_t stream) {
  const int*   tokens = (const int*)  d_in[0];
  const float* enc    = (const float*)d_in[1];
  // d_in[2] = tgt_mask (causal, computed analytically)
  const float* emb    = (const float*)d_in[3];
  const float* Wqkvo  = (const float*)d_in[4];   // [L,2,4,D,D]
  const float* bqkvo  = (const float*)d_in[5];   // [L,2,4,D]
  const float* W1     = (const float*)d_in[6];   // [L,D,DFF]
  const float* b1     = (const float*)d_in[7];
  const float* W2     = (const float*)d_in[8];   // [L,DFF,D]
  const float* b2     = (const float*)d_in[9];
  const float* ln_g   = (const float*)d_in[10];  // [L,3,D]
  const float* ln_b   = (const float*)d_in[11];
  const float* Wout   = (const float*)d_in[12];  // [D,V]
  const float* bout   = (const float*)d_in[13];

  float* out_logits = (float*)d_out;                               // [B,ST,V]
  float* self_attn  = out_logits + (size_t)B_ * ST_ * V_;          // [L,B,H,ST,ST]
  float* cross_attn = self_attn + (size_t)L_ * BHSS_;              // [L,B,H,ST,SS]

  // workspace carve-out
  char* wp = (char*)d_ws;
  auto carve = [&](size_t bytes) -> void* {
    void* r = (void*)wp; wp += (bytes + 255) & ~(size_t)255; return r;
  };
  const size_t ND = (size_t)NT_ * D_;
  float*  x_f32   = (float*) carve(ND * 4);
  float*  tmp_f32 = (float*) carve(ND * 4);
  __bf16* x_bf    = (__bf16*)carve(ND * 2);
  __bf16* enc_bf  = (__bf16*)carve(ND * 2);
  __bf16* q_bf    = (__bf16*)carve(ND * 2);
  __bf16* k_bf    = (__bf16*)carve(ND * 2);
  __bf16* v_bf    = (__bf16*)carve(ND * 2);
  __bf16* vt_bf   = (__bf16*)carve(ND * 2);
  __bf16* ctx_bf  = (__bf16*)carve(ND * 2);
  __bf16* ff_bf   = (__bf16*)carve((size_t)NT_ * DFF_ * 2);
  __bf16* attn_bf = (__bf16*)carve(BHSS_ * 2);
  __bf16* wqkvoT  = (__bf16*)carve((size_t)L_ * 2 * 4 * D_ * D_ * 2);
  __bf16* w1T     = (__bf16*)carve((size_t)L_ * D_ * DFF_ * 2);
  __bf16* w2T     = (__bf16*)carve((size_t)L_ * DFF_ * D_ * 2);
  __bf16* woutT   = (__bf16*)carve((size_t)D_ * V_ * 2);

  // ---- one-time per call: convert/transpose weights + encoder to bf16 ----
  {
    dim3 g((D_ * D_ + 255) / 256, L_ * 2 * 4);
    transpose_bf16_kernel<<<g, 256, 0, stream>>>(Wqkvo, wqkvoT, D_, D_);
  }
  {
    dim3 g((D_ * DFF_ + 255) / 256, L_);
    transpose_bf16_kernel<<<g, 256, 0, stream>>>(W1, w1T, D_, DFF_);
    transpose_bf16_kernel<<<g, 256, 0, stream>>>(W2, w2T, DFF_, D_);
  }
  {
    dim3 g((D_ * V_ + 255) / 256, 1);
    transpose_bf16_kernel<<<g, 256, 0, stream>>>(Wout, woutT, D_, V_);
  }
  to_bf16_kernel<<<(int)((ND + 255) / 256), 256, 0, stream>>>(enc, enc_bf, (int)ND);

  // ---- embedding + positional encoding ----
  embed_pe_kernel<<<(int)(ND / 256), 256, 0, stream>>>(tokens, emb, x_f32, x_bf);

  const long long sQb = (long long)ST_ * D_, sQh = DK_;       // [B,S,D] head views
  const long long sPb = (long long)H_ * ST_ * ST_, sPh = (long long)ST_ * ST_;
  const long long sVtb = (long long)H_ * DK_ * ST_, sVth = (long long)DK_ * ST_;

  for (int l = 0; l < L_; ++l) {
    // ================= self attention =================
    const __bf16* Wl = wqkvoT + (size_t)(l * 2 + 0) * 4 * D_ * D_;
    const float*  bl = bqkvo + (size_t)(l * 2 + 0) * 4 * D_;
    launch_gemm(stream, x_bf, D_, 0, 0, Wl + 0 * D_ * D_, D_, 0, 0,
                nullptr, q_bf, D_, 0, 0, bl + 0 * D_, NT_, D_, D_, 1, 1, 1.0f, 0);
    launch_gemm(stream, x_bf, D_, 0, 0, Wl + 1 * (size_t)D_ * D_, D_, 0, 0,
                nullptr, k_bf, D_, 0, 0, bl + 1 * D_, NT_, D_, D_, 1, 1, 1.0f, 0);
    launch_gemm(stream, x_bf, D_, 0, 0, Wl + 2 * (size_t)D_ * D_, D_, 0, 0,
                nullptr, v_bf, D_, 0, 0, bl + 2 * D_, NT_, D_, D_, 1, 1, 1.0f, 0);
    transpose_v_kernel<<<(int)(ND / 256), 256, 0, stream>>>(v_bf, vt_bf, ST_);

    float* sc = self_attn + (size_t)l * BHSS_;
    // scores = Q K^T / sqrt(dk)  (K rows serve directly as BT)
    launch_gemm(stream, q_bf, D_, sQb, sQh, k_bf, D_, sQb, sQh,
                sc, nullptr, ST_, sPb, sPh, nullptr,
                ST_, ST_, DK_, B_ * H_, H_, 0.125f, 0);
    softmax_kernel<<<B_ * H_ * ST_, 256, 0, stream>>>(sc, attn_bf, ST_, 1);
    // ctx = P V   (VT rows serve as BT)
    launch_gemm(stream, attn_bf, ST_, sPb, sPh, vt_bf, ST_, sVtb, sVth,
                nullptr, ctx_bf, D_, sQb, sQh, nullptr,
                ST_, DK_, ST_, B_ * H_, H_, 1.0f, 0);
    // output projection -> tmp, then residual + LN
    launch_gemm(stream, ctx_bf, D_, 0, 0, Wl + 3 * (size_t)D_ * D_, D_, 0, 0,
                tmp_f32, nullptr, D_, 0, 0, bl + 3 * D_, NT_, D_, D_, 1, 1, 1.0f, 0);
    add_ln_kernel<<<NT_, 256, 0, stream>>>(x_f32, x_bf, tmp_f32,
                                           ln_g + (size_t)(l * 3 + 0) * D_,
                                           ln_b + (size_t)(l * 3 + 0) * D_);

    // ================= cross attention =================
    const __bf16* Wc = wqkvoT + (size_t)(l * 2 + 1) * 4 * D_ * D_;
    const float*  bc = bqkvo + (size_t)(l * 2 + 1) * 4 * D_;
    launch_gemm(stream, x_bf, D_, 0, 0, Wc + 0 * D_ * D_, D_, 0, 0,
                nullptr, q_bf, D_, 0, 0, bc + 0 * D_, NT_, D_, D_, 1, 1, 1.0f, 0);
    launch_gemm(stream, enc_bf, D_, 0, 0, Wc + 1 * (size_t)D_ * D_, D_, 0, 0,
                nullptr, k_bf, D_, 0, 0, bc + 1 * D_, NT_, D_, D_, 1, 1, 1.0f, 0);
    launch_gemm(stream, enc_bf, D_, 0, 0, Wc + 2 * (size_t)D_ * D_, D_, 0, 0,
                nullptr, v_bf, D_, 0, 0, bc + 2 * D_, NT_, D_, D_, 1, 1, 1.0f, 0);
    transpose_v_kernel<<<(int)(ND / 256), 256, 0, stream>>>(v_bf, vt_bf, SS_);

    float* cc = cross_attn + (size_t)l * BHSS_;
    launch_gemm(stream, q_bf, D_, sQb, sQh, k_bf, D_, (long long)SS_ * D_, sQh,
                cc, nullptr, SS_, sPb, sPh, nullptr,
                ST_, SS_, DK_, B_ * H_, H_, 0.125f, 0);
    softmax_kernel<<<B_ * H_ * ST_, 256, 0, stream>>>(cc, attn_bf, SS_, 0);
    launch_gemm(stream, attn_bf, SS_, sPb, sPh, vt_bf, SS_, sVtb, sVth,
                nullptr, ctx_bf, D_, sQb, sQh, nullptr,
                ST_, DK_, SS_, B_ * H_, H_, 1.0f, 0);
    launch_gemm(stream, ctx_bf, D_, 0, 0, Wc + 3 * (size_t)D_ * D_, D_, 0, 0,
                tmp_f32, nullptr, D_, 0, 0, bc + 3 * D_, NT_, D_, D_, 1, 1, 1.0f, 0);
    add_ln_kernel<<<NT_, 256, 0, stream>>>(x_f32, x_bf, tmp_f32,
                                           ln_g + (size_t)(l * 3 + 1) * D_,
                                           ln_b + (size_t)(l * 3 + 1) * D_);

    // ================= feed forward =================
    launch_gemm(stream, x_bf, D_, 0, 0, w1T + (size_t)l * D_ * DFF_, D_, 0, 0,
                nullptr, ff_bf, DFF_, 0, 0, b1 + (size_t)l * DFF_,
                NT_, DFF_, D_, 1, 1, 1.0f, 1 /*relu*/);
    launch_gemm(stream, ff_bf, DFF_, 0, 0, w2T + (size_t)l * DFF_ * D_, DFF_, 0, 0,
                tmp_f32, nullptr, D_, 0, 0, b2 + (size_t)l * D_,
                NT_, D_, DFF_, 1, 1, 1.0f, 0);
    add_ln_kernel<<<NT_, 256, 0, stream>>>(x_f32, x_bf, tmp_f32,
                                           ln_g + (size_t)(l * 3 + 2) * D_,
                                           ln_b + (size_t)(l * 3 + 2) * D_);
  }

  // ================= vocab projection =================
  launch_gemm(stream, x_bf, D_, 0, 0, woutT, D_, 0, 0,
              out_logits, nullptr, V_, 0, 0, bout, NT_, V_, D_, 1, 1, 1.0f, 0);
}
